// MFA_12618613916389
// MI455X (gfx1250) — matrix operand order repaired and checked by reading, not compiled, not run on hardware
//
#include <hip/hip_runtime.h>
#include <math.h>

typedef _Float16 v16h __attribute__((ext_vector_type(16)));
typedef _Float16 v8h  __attribute__((ext_vector_type(8)));
typedef float    v8f  __attribute__((ext_vector_type(8)));

#define BATCH 16
#define NN 1024
#define DD 256
#define EPS 0.1f
#define INV_EPS 10.0f
#define MAX_ITER 100
#define THRESH 0.1f
#define LOG_MU (-6.9314616f)   /* log(1/1024 + 1e-8) */

// ---------------- zero init ----------------
__global__ void MFA_zero_kernel(float* __restrict__ p, int n) {
    int i = blockIdx.x * blockDim.x + threadIdx.x;
    if (i < n) p[i] = 0.0f;
}

// ---------------- normalize rows + convert to f16 ----------------
__global__ void MFA_nrm_kernel(const float* __restrict__ x, const float* __restrict__ y,
                               _Float16* __restrict__ xh, _Float16* __restrict__ yh) {
    __shared__ float red[256];
    int row = blockIdx.x;                       // 0 .. 2*BATCH*NN-1
    const float* src;
    _Float16* dst;
    if (row < BATCH * NN) { src = x + (size_t)row * DD;              dst = xh + (size_t)row * DD; }
    else                  { src = y + (size_t)(row - BATCH*NN) * DD; dst = yh + (size_t)(row - BATCH*NN) * DD; }
    int t = threadIdx.x;
    float v = src[t];
    red[t] = v * v;
    __syncthreads();
    for (int s = 128; s > 0; s >>= 1) {
        if (t < s) red[t] += red[t + s];
        __syncthreads();
    }
    float scale = 1.0f / fmaxf(sqrtf(red[0]), 1e-8f);
    dst[t] = (_Float16)(v * scale);
}

// ---------------- E = 1 - xn @ yn^T  via f16 WMMA, f32 accumulate ----------------
// block = 256 threads = 8 waves; block tile 128x128; wave tile 64(M) x 32(N) = 4x2 WMMA tiles
__global__ void MFA_gemm_kernel(const _Float16* __restrict__ xh, const _Float16* __restrict__ yh,
                                float* __restrict__ E) {
    int bb    = blockIdx.z;
    int tileM = blockIdx.y * 128;
    int tileN = blockIdx.x * 128;
    int tid  = threadIdx.x;
    int wave = tid >> 5, lane = tid & 31;
    int wm = wave >> 2, wn = wave & 3;
    int hlf = lane >> 4, mr = lane & 15;

    const _Float16* xb = xh + (size_t)bb * NN * DD;
    const _Float16* yb = yh + (size_t)bb * NN * DD;

    v8f acc[4][2];
    v8f zero = {0.f,0.f,0.f,0.f,0.f,0.f,0.f,0.f};
#pragma unroll
    for (int i = 0; i < 4; i++)
#pragma unroll
        for (int j = 0; j < 2; j++) acc[i][j] = zero;

    const _Float16* arow[4];
    const _Float16* brow[2];
#pragma unroll
    for (int sm = 0; sm < 4; sm++)
        arow[sm] = xb + (size_t)(tileM + wm*64 + sm*16 + mr) * DD;
#pragma unroll
    for (int sn = 0; sn < 2; sn++)
        brow[sn] = yb + (size_t)(tileN + wn*32 + sn*16 + mr) * DD;

    for (int k0 = 0; k0 < DD; k0 += 32) {
        v16h afr[4], bfr[2];
#pragma unroll
        for (int sm = 0; sm < 4; sm++) {
            // A layout: lanes 0-15 hold K = 0..7 then 16..23; lanes 16-31 hold K = 8..15 then 24..31
            v8h lo = *(const v8h*)(arow[sm] + k0 + hlf * 8);
            v8h hi = *(const v8h*)(arow[sm] + k0 + 16 + hlf * 8);
            afr[sm] = __builtin_shufflevector(lo, hi, 0,1,2,3,4,5,6,7,8,9,10,11,12,13,14,15);
        }
#pragma unroll
        for (int sn = 0; sn < 2; sn++) {
            // B layout: lanes 0-15 hold K = 0..7 then 16..23; lanes 16-31 hold K = 8..15 then 24..31 (col = lane%16)
            v8h c0 = *(const v8h*)(brow[sn] + k0 + hlf * 8);
            v8h c1 = *(const v8h*)(brow[sn] + k0 + 16 + hlf * 8);
            bfr[sn] = __builtin_shufflevector(c0, c1, 0,1,2,3,4,5,6,7,8,9,10,11,12,13,14,15);
        }
#pragma unroll
        for (int sm = 0; sm < 4; sm++)
#pragma unroll
            for (int sn = 0; sn < 2; sn++)
                acc[sm][sn] = __builtin_amdgcn_wmma_f32_16x16x32_f16(
                    false, afr[sm], false, bfr[sn], (short)0, acc[sm][sn], false, false);
    }

    // C layout: VGPR r -> M = r + 8*(lane/16), N = lane%16
#pragma unroll
    for (int sm = 0; sm < 4; sm++) {
        int m0 = tileM + wm*64 + sm*16 + hlf*8;
#pragma unroll
        for (int sn = 0; sn < 2; sn++) {
            int col = tileN + wn*32 + sn*16 + mr;
#pragma unroll
            for (int r = 0; r < 8; r++)
                E[((size_t)bb*NN + m0 + r) * NN + col] = 1.0f - acc[sm][sn][r];
        }
    }
}

// ---------------- Sinkhorn row update (u) + err accumulation ----------------
__global__ void MFA_row_kernel(const float* __restrict__ E, float* __restrict__ u,
                               const float* __restrict__ v, float* __restrict__ err,
                               const int* __restrict__ done) {
    if (*done) return;
    __shared__ float smx[256], ssm[256];
    int i = blockIdx.x, bb = blockIdx.y, t = threadIdx.x;
    const float* Erow = E + ((size_t)bb*NN + i) * NN;
    const float* vb = v + bb*NN;
    float ui = u[bb*NN + i];
    float m = -INFINITY, s = 0.0f;
    for (int j = t; j < NN; j += 256) {
        float xv = (ui + vb[j] - Erow[j]) * INV_EPS;
        if (xv > m) { s = s * __expf(m - xv) + 1.0f; m = xv; }
        else        { s += __expf(xv - m); }
    }
    smx[t] = m; ssm[t] = s;
    __syncthreads();
    for (int st = 128; st > 0; st >>= 1) {
        if (t < st) {
            float m1 = smx[t], s1 = ssm[t];
            float m2 = smx[t+st], s2 = ssm[t+st];
            float mm = fmaxf(m1, m2);
            smx[t] = mm;
            ssm[t] = s1 * __expf(m1 - mm) + s2 * __expf(m2 - mm);
        }
        __syncthreads();
    }
    if (t == 0) {
        float lse = smx[0] + __logf(ssm[0]);
        float un  = EPS * (LOG_MU - lse) + ui;
        u[bb*NN + i] = un;
        atomicAdd(err, fabsf(un - ui));
    }
}

// ---------------- Sinkhorn column update (v) ----------------
// block = 32 columns x 8 row-groups -> coalesced 128B reads per row step
__global__ void MFA_col_kernel(const float* __restrict__ E, const float* __restrict__ u,
                               float* __restrict__ v, const int* __restrict__ done) {
    if (*done) return;
    __shared__ float smx[256], ssm[256];
    int bb = blockIdx.y;
    int c = threadIdx.x & 31, g = threadIdx.x >> 5;
    int j = blockIdx.x * 32 + c;
    const float* ub = u + bb*NN;
    float vj = v[bb*NN + j];
    float m = -INFINITY, s = 0.0f;
    for (int i = g; i < NN; i += 8) {
        float xv = (ub[i] + vj - E[((size_t)bb*NN + i)*NN + j]) * INV_EPS;
        if (xv > m) { s = s * __expf(m - xv) + 1.0f; m = xv; }
        else        { s += __expf(xv - m); }
    }
    smx[threadIdx.x] = m; ssm[threadIdx.x] = s;
    __syncthreads();
    if (g == 0) {
        for (int gg = 1; gg < 8; gg++) {
            float m2 = smx[gg*32 + c], s2 = ssm[gg*32 + c];
            float mm = fmaxf(m, m2);
            s = s * __expf(m - mm) + s2 * __expf(m2 - mm);
            m = mm;
        }
        float lse = m + __logf(s);
        v[bb*NN + j] = EPS * (LOG_MU - lse) + vj;   // log_nu == log_mu (N == M)
    }
}

// ---------------- convergence flag ----------------
__global__ void MFA_done_kernel(const float* __restrict__ err, int* __restrict__ done) {
    if (*done) return;
    if (*err < THRESH * (float)BATCH) *done = 1;   // err_ref = sum/BATCH < THRESH
}

// ---------------- final cost: sum(exp(M) * E) ----------------
__global__ void MFA_cost_kernel(const float* __restrict__ E, const float* __restrict__ u,
                                const float* __restrict__ v, float* __restrict__ out) {
    __shared__ float red[256];
    int i = blockIdx.x, bb = blockIdx.y, t = threadIdx.x;
    const float* Erow = E + ((size_t)bb*NN + i) * NN;
    const float* vb = v + bb*NN;
    float ui = u[bb*NN + i];
    float acc = 0.0f;
    for (int j = t; j < NN; j += 256) {
        float e = Erow[j];
        acc += __expf((ui + vb[j] - e) * INV_EPS) * e;
    }
    red[t] = acc;
    __syncthreads();
    for (int s = 128; s > 0; s >>= 1) {
        if (t < s) red[t] += red[t + s];
        __syncthreads();
    }
    if (t == 0) atomicAdd(out + bb, red[0]);
}

extern "C" void kernel_launch(void* const* d_in, const int* in_sizes, int n_in,
                              void* d_out, int out_size, void* d_ws, size_t ws_size,
                              hipStream_t stream) {
    const float* x = (const float*)d_in[0];
    const float* y = (const float*)d_in[1];
    float* out = (float*)d_out;

    char* ws = (char*)d_ws;
    size_t o = 0;
    _Float16* xh = (_Float16*)(ws + o); o += (size_t)BATCH*NN*DD*2;   // 8 MB
    _Float16* yh = (_Float16*)(ws + o); o += (size_t)BATCH*NN*DD*2;   // 8 MB
    float*    E  = (float*)(ws + o);    o += (size_t)BATCH*NN*NN*4;   // 64 MB (fits in 192MB L2)
    float*    u  = (float*)(ws + o);    o += (size_t)BATCH*NN*4;
    float*    v  = (float*)(ws + o);    o += (size_t)BATCH*NN*4;
    float*    err= (float*)(ws + o);    o += (size_t)MAX_ITER*4;
    int*     done= (int*)(ws + o);      o += 256;

    // zero u, v, err, done (contiguous floats) and the output (graph-replay determinism)
    int miscN = BATCH*NN*2 + MAX_ITER + 64;
    MFA_zero_kernel<<<(miscN + 255)/256, 256, 0, stream>>>(u, miscN);
    MFA_zero_kernel<<<1, 32, 0, stream>>>(out, out_size);

    MFA_nrm_kernel<<<2*BATCH*NN, 256, 0, stream>>>(x, y, xh, yh);
    MFA_gemm_kernel<<<dim3(NN/128, NN/128, BATCH), 256, 0, stream>>>(xh, yh, E);

    for (int it = 0; it < MAX_ITER; it++) {
        MFA_row_kernel<<<dim3(NN, BATCH), 256, 0, stream>>>(E, u, v, err + it, done);
        MFA_col_kernel<<<dim3(NN/32, BATCH), 256, 0, stream>>>(E, u, v, done);
        MFA_done_kernel<<<1, 1, 0, stream>>>(err + it, done);
    }

    MFA_cost_kernel<<<dim3(NN, BATCH), 256, 0, stream>>>(E, u, v, out);
}
